// PriorKNN_3341484556728
// MI455X (gfx1250) — matrix operand order repaired and checked by reading, not compiled, not run on hardware
//
#include <hip/hip_runtime.h>
#include <hip/hip_bf16.h>
#include <math.h>

typedef __attribute__((ext_vector_type(16))) __bf16 v16bf;
typedef __attribute__((ext_vector_type(8)))  float  v8f;

#define BATCH 256
#define MDIM  306      // measurement dim
#define NDIM  984      // sparse code dim
#define NPAD  1024     // NDIM padded to 32
#define MPAD  320      // MDIM padded to 32
#define KT1   32       // NPAD/32  (GEMM1 k-steps)
#define NT1   20       // MPAD/16  (GEMM1 n-tiles)
#define KT2   10       // MPAD/32  (GEMM2 k-steps)
#define NT2   64       // NPAD/16  (GEMM2 n-tiles)

__device__ __forceinline__ v8f wmma_bf16(v16bf a, v16bf b, v8f c) {
  return __builtin_amdgcn_wmma_f32_16x16x32_bf16(false, a, false, b, (short)0, c, false, false);
}

// -------------------------------------------------------------------------
// Spectral norm of phi (complex) via power iteration on A = phi^H phi.
// step = 1 / s0^2 = 1 / lambda_max(A).
// -------------------------------------------------------------------------
__global__ void k_spectral(const float* __restrict__ phre, const float* __restrict__ phim,
                           float* __restrict__ step_out) {
  __shared__ float vr[NDIM], vi[NDIM], ur[MDIM], ui[MDIM], red[512];
  const int tid = threadIdx.x, nth = blockDim.x;
  for (int n = tid; n < NDIM; n += nth) { vr[n] = 1.0f; vi[n] = 0.0f; }
  __syncthreads();
  float lam = 1.0f;
  for (int it = 0; it < 30; ++it) {
    for (int m = tid; m < MDIM; m += nth) {
      const float* pr = phre + (size_t)m * NDIM;
      const float* pi = phim + (size_t)m * NDIM;
      float ar = 0.f, ai = 0.f;
      for (int n = 0; n < NDIM; ++n) {
        float br = pr[n], bi = pi[n];
        ar += br * vr[n] - bi * vi[n];
        ai += br * vi[n] + bi * vr[n];
      }
      ur[m] = ar; ui[m] = ai;
    }
    __syncthreads();
    float part = 0.f;
    for (int n = tid; n < NDIM; n += nth) {
      float ar = 0.f, ai = 0.f;
      for (int m = 0; m < MDIM; ++m) {
        float br = phre[(size_t)m * NDIM + n], bi = phim[(size_t)m * NDIM + n];
        ar += br * ur[m] + bi * ui[m];     // conj(phi) * u
        ai += br * ui[m] - bi * ur[m];
      }
      vr[n] = ar; vi[n] = ai;
      part += ar * ar + ai * ai;
    }
    red[tid] = part;
    __syncthreads();
    for (int s = nth >> 1; s > 0; s >>= 1) { if (tid < s) red[tid] += red[tid + s]; __syncthreads(); }
    lam = sqrtf(red[0]);
    float inv = (lam > 0.f) ? 1.0f / lam : 0.0f;
    __syncthreads();
    for (int n = tid; n < NDIM; n += nth) { vr[n] *= inv; vi[n] *= inv; }
    __syncthreads();
  }
  if (tid == 0) step_out[0] = (lam > 0.f) ? 1.0f / lam : 1.0f;
}

// -------------------------------------------------------------------------
// Pre-swizzle phi^T (GEMM1 B) and conj(phi) (GEMM2 B) into bf16 WMMA-B
// fragment layout: frag tile = 512 bf16, lane-major (lane*16 + j);
// element (lane,j) -> K = (lane>=16 ? 8:0)+(j<8 ? j : 8+j), N = lane&15.
// -------------------------------------------------------------------------
__global__ void k_prep_frags(const float* __restrict__ phre, const float* __restrict__ phim,
                             __bf16* __restrict__ bf1r, __bf16* __restrict__ bf1i,
                             __bf16* __restrict__ bf2r, __bf16* __restrict__ bf2i) {
  const int idx = blockIdx.x * blockDim.x + threadIdx.x;
  const int T1 = KT1 * NT1 * 512;
  const int T2 = KT2 * NT2 * 512;
  if (idx < T1) {
    int kt = idx / (NT1 * 512); int rem = idx % (NT1 * 512);
    int nt = rem / 512;  int e = rem % 512;
    int lane = e >> 4, j = e & 15;
    int k = kt * 32 + ((lane & 16) ? 8 : 0) + ((j < 8) ? j : 8 + j);   // over NPAD
    int n = nt * 16 + (lane & 15);                  // over MPAD
    float re = 0.f, im = 0.f;
    if (k < NDIM && n < MDIM) { re = phre[(size_t)n * NDIM + k]; im = phim[(size_t)n * NDIM + k]; }
    bf1r[idx] = (__bf16)re; bf1i[idx] = (__bf16)im;
  } else if (idx < T1 + T2) {
    int id2 = idx - T1;
    int kt = id2 / (NT2 * 512); int rem = id2 % (NT2 * 512);
    int nt = rem / 512;  int e = rem % 512;
    int lane = e >> 4, j = e & 15;
    int k = kt * 32 + ((lane & 16) ? 8 : 0) + ((j < 8) ? j : 8 + j);   // over MPAD
    int n = nt * 16 + (lane & 15);                  // over NPAD
    float re = 0.f, im = 0.f;
    if (k < MDIM && n < NDIM) { re = phre[(size_t)k * NDIM + n]; im = -phim[(size_t)k * NDIM + n]; }
    bf2r[id2] = (__bf16)re; bf2i[id2] = (__bf16)im;
  }
}

// -------------------------------------------------------------------------
// Persistent fused ISTA: 16 workgroups x 16 rows, 50 iterations in-kernel.
// Wave index is readfirstlane'd so all wave-level tile decisions are scalar
// branches (WMMA must never sit under an EXEC mask).
// -------------------------------------------------------------------------
__global__ void __launch_bounds__(512, 1)
k_ista(const float* __restrict__ rre, const float* __restrict__ rim,
       const __bf16* __restrict__ bf1r, const __bf16* __restrict__ bf1i,
       const __bf16* __restrict__ bf2r, const __bf16* __restrict__ bf2i,
       const float* __restrict__ step_ptr, float* __restrict__ hout) {
  extern __shared__ char dynsmem[];
  float*  wr = (float*)dynsmem;              // [16][NPAD]
  float*  wi = wr + 16 * NPAD;
  float*  rr = wi + 16 * NPAD;               // [16][MPAD]
  float*  ri = rr + 16 * MPAD;
  __bf16* er = (__bf16*)(ri + 16 * MPAD);    // [16][MPAD]
  __bf16* ei = er + 16 * MPAD;

  const int tid = threadIdx.x, nth = blockDim.x;
  const int lane = tid & 31;
  const int wv = __builtin_amdgcn_readfirstlane(tid >> 5);   // scalar wave id
  const int row0 = blockIdx.x * 16;

  for (int i = tid; i < 16 * NPAD; i += nth) { wr[i] = 0.f; wi[i] = 0.f; }
  for (int i = tid; i < 16 * MPAD; i += nth) {
    int m = i / MPAD, n = i % MPAD;
    float a = 0.f, b = 0.f;
    if (n < MDIM) { a = rre[(size_t)(row0 + m) * MDIM + n]; b = rim[(size_t)(row0 + m) * MDIM + n]; }
    rr[i] = a; ri[i] = b;
  }
  __syncthreads();
  const float step = step_ptr[0];
  const float thr  = 0.1f * step;
  const int arow = lane & 15;
  const int asel = (lane & 16) ? 8 : 0;
  const int bcol = lane & 15;
  const v8f VZ = {0,0,0,0,0,0,0,0};

  for (int it = 0; it < 50; ++it) {
    // ---- GEMM1: err = w @ phi^T - r   (K=NPAD, N=MPAD); up to 2 n-tiles/wave ----
    {
      const int ntA = wv, ntB = wv + 16;
      const bool two = (ntB < NT1);                 // scalar condition
      v8f cre0 = VZ, cim0 = VZ, cre1 = VZ, cim1 = VZ;
      for (int kt = 0; kt < KT1; ++kt) {
        const float* pwr = wr + arow * NPAD + kt * 32 + asel;
        const float* pwi = wi + arow * NPAD + kt * 32 + asel;
        v16bf Ar, Ai, Ain;
        #pragma unroll
        for (int j = 0; j < 8; ++j) {
          float xr = pwr[j], yr = pwr[16 + j];
          float xi = pwi[j], yi = pwi[16 + j];
          Ar[j] = (__bf16)xr;    Ar[8 + j] = (__bf16)yr;
          Ai[j] = (__bf16)xi;    Ai[8 + j] = (__bf16)yi;
          Ain[j] = (__bf16)(-xi); Ain[8 + j] = (__bf16)(-yi);
        }
        const __bf16* brow = bf1r + ((size_t)kt * NT1) * 512 + lane * 16;
        const __bf16* irow = bf1i + ((size_t)kt * NT1) * 512 + lane * 16;
        if (kt + 1 < KT1) {
          __builtin_prefetch(brow + (size_t)NT1 * 512 + ntA * 512, 0, 1);
          __builtin_prefetch(irow + (size_t)NT1 * 512 + ntA * 512, 0, 1);
        }
        v16bf Br = *(const v16bf*)(brow + ntA * 512);
        v16bf Bi = *(const v16bf*)(irow + ntA * 512);
        cre0 = wmma_bf16(Ar,  Br, cre0);
        cre0 = wmma_bf16(Ain, Bi, cre0);
        cim0 = wmma_bf16(Ar,  Bi, cim0);
        cim0 = wmma_bf16(Ai,  Br, cim0);
        if (two) {
          v16bf Br2 = *(const v16bf*)(brow + ntB * 512);
          v16bf Bi2 = *(const v16bf*)(irow + ntB * 512);
          cre1 = wmma_bf16(Ar,  Br2, cre1);
          cre1 = wmma_bf16(Ain, Bi2, cre1);
          cim1 = wmma_bf16(Ar,  Bi2, cim1);
          cim1 = wmma_bf16(Ai,  Br2, cim1);
        }
      }
      #pragma unroll
      for (int rI = 0; rI < 8; ++rI) {
        int m = rI + asel;
        int n = ntA * 16 + bcol;
        er[m * MPAD + n] = (__bf16)(cre0[rI] - rr[m * MPAD + n]);
        ei[m * MPAD + n] = (__bf16)(cim0[rI] - ri[m * MPAD + n]);
      }
      if (two) {
        #pragma unroll
        for (int rI = 0; rI < 8; ++rI) {
          int m = rI + asel;
          int n = ntB * 16 + bcol;
          er[m * MPAD + n] = (__bf16)(cre1[rI] - rr[m * MPAD + n]);
          ei[m * MPAD + n] = (__bf16)(cim1[rI] - ri[m * MPAD + n]);
        }
      }
    }
    __syncthreads();
    // ---- GEMM2: w -= step * (err @ conj(phi)); 4 n-tiles per wave ----
    {
      v8f cr0 = VZ, cr1 = VZ, cr2 = VZ, cr3 = VZ;
      v8f ci0 = VZ, ci1 = VZ, ci2 = VZ, ci3 = VZ;
      for (int kt = 0; kt < KT2; ++kt) {
        const __bf16* per = er + arow * MPAD + kt * 32 + asel;
        const __bf16* pei = ei + arow * MPAD + kt * 32 + asel;
        v16bf Er, Ei, Ein;
        #pragma unroll
        for (int j = 0; j < 8; ++j) {
          float xr = (float)per[j], yr = (float)per[16 + j];
          float xi = (float)pei[j], yi = (float)pei[16 + j];
          Er[j] = (__bf16)xr;    Er[8 + j] = (__bf16)yr;
          Ei[j] = (__bf16)xi;    Ei[8 + j] = (__bf16)yi;
          Ein[j] = (__bf16)(-xi); Ein[8 + j] = (__bf16)(-yi);
        }
        const __bf16* brow = bf2r + ((size_t)kt * NT2) * 512 + lane * 16;
        const __bf16* irow = bf2i + ((size_t)kt * NT2) * 512 + lane * 16;
        #pragma unroll
        for (int t = 0; t < 4; ++t) {
          int nt = wv + t * 16;                     // scalar tile index
          v16bf Br = *(const v16bf*)(brow + (size_t)nt * 512);
          v16bf Bi = *(const v16bf*)(irow + (size_t)nt * 512);
          v8f* pcr = (t == 0) ? &cr0 : (t == 1) ? &cr1 : (t == 2) ? &cr2 : &cr3;
          v8f* pci = (t == 0) ? &ci0 : (t == 1) ? &ci1 : (t == 2) ? &ci2 : &ci3;
          *pcr = wmma_bf16(Er,  Br, *pcr);
          *pcr = wmma_bf16(Ein, Bi, *pcr);
          *pci = wmma_bf16(Er,  Bi, *pci);
          *pci = wmma_bf16(Ei,  Br, *pci);
        }
      }
      #pragma unroll
      for (int t = 0; t < 4; ++t) {
        int nt = wv + t * 16;
        v8f cre = (t == 0) ? cr0 : (t == 1) ? cr1 : (t == 2) ? cr2 : cr3;
        v8f cim = (t == 0) ? ci0 : (t == 1) ? ci1 : (t == 2) ? ci2 : ci3;
        #pragma unroll
        for (int rI = 0; rI < 8; ++rI) {
          int m = rI + asel;
          int n = nt * 16 + bcol;
          float a = wr[m * NPAD + n] - step * cre[rI];
          float b = wi[m * NPAD + n] - step * cim[rI];
          float mag = sqrtf(a * a + b * b);
          float sc = fmaxf(mag - thr, 0.f) / ((mag > 0.f) ? mag : 1.f);
          wr[m * NPAD + n] = a * sc;
          wi[m * NPAD + n] = b * sc;
        }
      }
    }
    __syncthreads();
  }
  for (int i = tid; i < 16 * NDIM; i += nth) {
    int m = i / NDIM, n = i % NDIM;
    float a = wr[m * NPAD + n], b = wi[m * NPAD + n];
    hout[(size_t)(row0 + m) * NDIM + n] = sqrtf(a * a + b * b);
  }
}

// -------------------------------------------------------------------------
// Implicit-GEMM 1D conv. Block = 8 waves; block owns one N-tile (16 output
// channels). The 16-row weight slice is staged into LDS once per block via
// CDNA5 async-to-LDS loads, then every wave amortizes one B fragment over
// 4 M-tiles. KSZ is a compile-time constant.
// -------------------------------------------------------------------------
template <int KSZ>
__global__ void k_conv(const float* __restrict__ x, const float* __restrict__ w,
                       const float* __restrict__ bias, float* __restrict__ y,
                       int Cin, int Lin, int Cout, int Lout, int stride) {
  extern __shared__ char dynsmem[];
  float* wsl = (float*)dynsmem;               // [16][Kpad]
  const int pad = KSZ / 2;
  const int K = Cin * KSZ;
  const int ktiles = (K + 31) >> 5;
  const int Kpad = ktiles * 32;
  const int tid = threadIdx.x;
  const int lane = tid & 31;
  const int wv = __builtin_amdgcn_readfirstlane(tid >> 5);   // scalar wave id
  const int tilesN = (Cout + 15) >> 4;
  const int Mtot = BATCH * Lout;
  const int tilesM = (Mtot + 15) >> 4;
  const int nt = blockIdx.x % tilesN;
  const int mg = blockIdx.x / tilesN;

  // zero pad region + rows beyond Cout
  for (int i = tid; i < 16 * Kpad; i += blockDim.x) {
    int row = i / Kpad, kk = i % Kpad;
    if (kk >= K || (nt * 16 + row) >= Cout) wsl[i] = 0.f;
  }
  // async-copy the valid weight slice (16B chunks), then tail elements
  {
    int chunksPerRow = K >> 2;
    int totalChunks = 16 * chunksPerRow;
    for (int c = tid; c < totalChunks; c += blockDim.x) {
      int row = c / chunksPerRow, kk4 = (c % chunksPerRow) << 2;
      int co = nt * 16 + row;
      if (co < Cout) {
        unsigned ldsOff = (unsigned)(size_t)&wsl[row * Kpad + kk4];
        const float* g = w + (size_t)co * K + kk4;
        asm volatile("global_load_async_to_lds_b128 %0, %1, off"
                     :: "v"(ldsOff), "v"(g) : "memory");
      }
    }
    int kk0 = chunksPerRow << 2;
    int rem = K - kk0;
    if (rem > 0) {
      for (int c = tid; c < 16 * rem; c += blockDim.x) {
        int row = c / rem, e = c % rem;
        int co = nt * 16 + row;
        if (co < Cout) wsl[row * Kpad + kk0 + e] = w[(size_t)co * K + kk0 + e];
      }
    }
    asm volatile("s_wait_asynccnt 0" ::: "memory");
  }
  __syncthreads();

  const int arow = lane & 15, asel = (lane & 16) ? 8 : 0, bcol = lane & 15;
  const int co = nt * 16 + bcol;
  const bool nok = co < Cout;
  const int mt0 = (mg * 8 + wv) * 4;          // scalar
  if (mt0 >= tilesM) return;                  // scalar whole-wave exit (no barriers after)

  int mrow[4], bI[4], tI[4];
  bool mok[4];
  #pragma unroll
  for (int t = 0; t < 4; ++t) {
    mrow[t] = (mt0 + t) * 16 + arow;
    mok[t]  = mrow[t] < Mtot;
    int mm  = mok[t] ? mrow[t] : 0;
    bI[t] = mm / Lout;  tI[t] = mm % Lout;
  }
  const v8f VZ = {0,0,0,0,0,0,0,0};
  v8f a0 = VZ, a1 = VZ, a2 = VZ, a3 = VZ;

  for (int kt = 0; kt < ktiles; ++kt) {
    v16bf Bv;
    const float* pw = &wsl[bcol * Kpad + kt * 32 + ((lane & 16) ? 16 : 0)];
    #pragma unroll
    for (int j = 0; j < 16; ++j) Bv[j] = (__bf16)pw[j];
    #pragma unroll
    for (int t = 0; t < 4; ++t) {
      v16bf A;
      #pragma unroll
      for (int j = 0; j < 16; ++j) {
        int kk = kt * 32 + 2 * asel + j;
        float v = 0.f;
        if (mok[t] && kk < K) {
          int ci = kk / KSZ, tap = kk % KSZ;
          int pos = tI[t] * stride + tap - pad;
          if (pos >= 0 && pos < Lin) v = x[((size_t)bI[t] * Cin + ci) * Lin + pos];
        }
        A[j] = (__bf16)v;
      }
      v8f* pa = (t == 0) ? &a0 : (t == 1) ? &a1 : (t == 2) ? &a2 : &a3;
      *pa = wmma_bf16(A, Bv, *pa);
    }
  }
  #pragma unroll
  for (int t = 0; t < 4; ++t) {
    v8f acc = (t == 0) ? a0 : (t == 1) ? a1 : (t == 2) ? a2 : a3;
    #pragma unroll
    for (int rI = 0; rI < 8; ++rI) {
      int mm = (mt0 + t) * 16 + rI + asel;
      if (mm < Mtot && nok) {
        int bb = mm / Lout, tt = mm % Lout;
        y[((size_t)bb * Cout + co) * Lout + tt] = acc[rI] + bias[co];
      }
    }
  }
}

__global__ void k_zero(float* p, int n) {
  int i = blockIdx.x * blockDim.x + threadIdx.x;
  if (i < n) p[i] = 0.f;
}

// Per-channel sum / sumsq over (B, L) with LDS pre-reduction.
__global__ void k_stats(const float* __restrict__ x, float* __restrict__ sum,
                        float* __restrict__ sumsq, int C, int L) {
  __shared__ float s1[256], s2[256];
  const int tid = threadIdx.x;
  for (int c = tid; c < C; c += blockDim.x) { s1[c] = 0.f; s2[c] = 0.f; }
  __syncthreads();
  size_t total = (size_t)BATCH * C * L;
  for (size_t i = blockIdx.x * (size_t)blockDim.x + tid; i < total;
       i += (size_t)gridDim.x * blockDim.x) {
    int c = (int)((i / (size_t)L) % (size_t)C);
    float v = x[i];
    atomicAdd(&s1[c], v);
    atomicAdd(&s2[c], v * v);
  }
  __syncthreads();
  for (int c = tid; c < C; c += blockDim.x) {
    atomicAdd(&sum[c], s1[c]);
    atomicAdd(&sumsq[c], s2[c]);
  }
}

// out = [relu]( bn(a; s1a..) [+ bn(b2; s1b..)] )
__global__ void k_bn_apply(const float* __restrict__ a, const float* s1a, const float* s2a,
                           const float* ga, const float* ba,
                           const float* __restrict__ b2, const float* s1b, const float* s2b,
                           const float* gb, const float* bb,
                           float* __restrict__ out, int C, int L, int relu) {
  size_t total = (size_t)BATCH * C * L;
  float invc = 1.0f / (float)(BATCH * L);
  for (size_t i = blockIdx.x * (size_t)blockDim.x + threadIdx.x; i < total;
       i += (size_t)gridDim.x * blockDim.x) {
    int c = (int)((i / (size_t)L) % (size_t)C);
    float m = s1a[c] * invc;
    float v = s2a[c] * invc - m * m;
    float r = (a[i] - m) * rsqrtf(v + 1e-5f) * ga[c] + ba[c];
    if (b2) {
      float m2 = s1b[c] * invc;
      float v2 = s2b[c] * invc - m2 * m2;
      r += (b2[i] - m2) * rsqrtf(v2 + 1e-5f) * gb[c] + bb[c];
    }
    if (relu) r = fmaxf(r, 0.f);
    out[i] = r;
  }
}

__global__ void k_pool(const float* __restrict__ x, float* __restrict__ out, int C, int L) {
  int i = blockIdx.x * blockDim.x + threadIdx.x;
  if (i >= BATCH * C) return;
  int b = i / C, c = i % C;
  const float* p = x + ((size_t)b * C + c) * L;
  float s = 0.f;
  for (int t = 0; t < L; ++t) s += p[t];
  out[i] = s / (float)L;
}

// C = X(MxK) * W(NxK)^T + bias ; wave per 16x16 tile; K multiple of 32.
__global__ void k_gemm(const float* __restrict__ X, const float* __restrict__ W,
                       const float* __restrict__ bias, float* __restrict__ Co,
                       int M, int N, int K, int relu) {
  const int lane = threadIdx.x & 31;
  const int wvL = __builtin_amdgcn_readfirstlane(threadIdx.x >> 5);  // scalar
  const int wvG = blockIdx.x * (blockDim.x >> 5) + wvL;
  const int tilesN = (N + 15) >> 4, tilesM = (M + 15) >> 4;
  if (wvG >= tilesM * tilesN) return;         // scalar whole-wave exit
  const int mt = wvG / tilesN, nt = wvG % tilesN;
  const int arow = lane & 15, asel = (lane & 16) ? 8 : 0, bcol = lane & 15;
  const int mrow = mt * 16 + arow; const bool mok = mrow < M;
  const int ncol = nt * 16 + bcol; const bool nok = ncol < N;
  v8f acc = {0,0,0,0,0,0,0,0};
  const int ktiles = K >> 5;
  for (int kt = 0; kt < ktiles; ++kt) {
    v16bf A, Bv;
    const float* px = X + (size_t)mrow * K + kt * 32 + 2 * asel;
    #pragma unroll
    for (int j = 0; j < 8; ++j) {
      float x0 = mok ? px[j] : 0.f;
      float x1 = mok ? px[8 + j] : 0.f;
      A[j] = (__bf16)x0; A[8 + j] = (__bf16)x1;
    }
    const float* pw = W + (size_t)ncol * K + kt * 32 + ((lane & 16) ? 16 : 0);
    #pragma unroll
    for (int j = 0; j < 16; ++j) { float v = nok ? pw[j] : 0.f; Bv[j] = (__bf16)v; }
    acc = wmma_bf16(A, Bv, acc);
  }
  #pragma unroll
  for (int rI = 0; rI < 8; ++rI) {
    int mm = mt * 16 + rI + asel;
    if (mm < M && nok) {
      float v = acc[rI] + (bias ? bias[ncol] : 0.f);
      if (relu) v = fmaxf(v, 0.f);
      Co[(size_t)mm * N + ncol] = v;
    }
  }
}

__global__ void k_l2norm(const float* __restrict__ x, float* __restrict__ out, int D) {
  __shared__ float red[256];
  const int row = blockIdx.x, tid = threadIdx.x;
  const float* p = x + (size_t)row * D;
  float s = 0.f;
  for (int j = tid; j < D; j += blockDim.x) { float v = p[j]; s += v * v; }
  red[tid] = s;
  __syncthreads();
  for (int st = blockDim.x >> 1; st > 0; st >>= 1) {
    if (tid < st) red[tid] += red[tid + st];
    __syncthreads();
  }
  float inv = 1.0f / fmaxf(sqrtf(red[0]), 1e-12f);
  for (int j = tid; j < D; j += blockDim.x) out[(size_t)row * D + j] = p[j] * inv;
}

// -------------------------------------------------------------------------
// Host driver
// -------------------------------------------------------------------------
struct BlkP { const float *w1,*b1,*g1,*be1,*w2,*b2,*g2,*be2,*ws,*bs,*gs,*bes; };
struct AllP {
  const float *r_real,*r_imag,*phi_re,*phi_im;
  BlkP b1, b2, b3;
  const float *fc_w,*fc_b,*cls_w,*cls_b,*p1_w,*p1_b,*p2_w,*p2_b;
};

static constexpr size_t SZB_F1 = (size_t)KT1 * NT1 * 512 * 2;
static constexpr size_t SZB_F2 = (size_t)KT2 * NT2 * 512 * 2;
static constexpr size_t O_BF1R = 256;
static constexpr size_t O_BF1I = O_BF1R + SZB_F1;
static constexpr size_t O_BF2R = O_BF1I + SZB_F1;
static constexpr size_t O_BF2I = O_BF2R + SZB_F2;
static constexpr size_t O_H    = O_BF2I + SZB_F2;
static constexpr size_t SZB_H  = (size_t)BATCH * NDIM * 4;
static constexpr size_t O_BUF1 = O_H + SZB_H;
static constexpr size_t SZB_BIG = (size_t)BATCH * 64 * 984 * 4;
static constexpr size_t O_BUF2 = O_BUF1 + SZB_BIG;
static constexpr size_t O_BUF3 = O_BUF2 + SZB_BIG;
static constexpr size_t O_STAT = O_BUF3 + SZB_BIG;
static constexpr size_t O_POOL = O_STAT + 4 * 256 * 4;
static constexpr size_t O_Q    = O_POOL + (size_t)BATCH * 256 * 4;
static constexpr size_t O_Z1   = O_Q + (size_t)BATCH * 1024 * 4;
static constexpr size_t O_Z2   = O_Z1 + (size_t)BATCH * 512 * 4;

extern "C" void kernel_launch(void* const* d_in, const int* in_sizes, int n_in,
                              void* d_out, int out_size, void* d_ws, size_t ws_size,
                              hipStream_t stream) {
  auto F = [&](int i) { return (const float*)d_in[i]; };
  AllP P;
  if (n_in > 0 && in_sizes[0] == BATCH * MDIM) {
    // insertion-order flatten
    P.r_real = F(0); P.r_imag = F(1); P.phi_re = F(2); P.phi_im = F(3);
    auto blk = [&](int base) {
      BlkP o;
      o.w1 = F(base + 0); o.b1 = F(base + 1); o.g1 = F(base + 2); o.be1 = F(base + 3);
      o.w2 = F(base + 4); o.b2 = F(base + 5); o.g2 = F(base + 6); o.be2 = F(base + 7);
      o.ws = F(base + 8); o.bs = F(base + 9); o.gs = F(base + 10); o.bes = F(base + 11);
      return o;
    };
    P.b1 = blk(4); P.b2 = blk(16); P.b3 = blk(28);
    P.fc_w = F(40); P.fc_b = F(41); P.cls_w = F(42); P.cls_b = F(43);
    P.p1_w = F(44); P.p1_b = F(45); P.p2_w = F(46); P.p2_b = F(47);
  } else {
    // jax sorted-key tree flatten
    auto blk = [&](int base) {
      BlkP o;
      o.b1 = F(base + 0); o.b2 = F(base + 1); o.be1 = F(base + 2); o.be2 = F(base + 3);
      o.bes = F(base + 4); o.bs = F(base + 5); o.g1 = F(base + 6); o.g2 = F(base + 7);
      o.gs = F(base + 8); o.w1 = F(base + 9); o.w2 = F(base + 10); o.ws = F(base + 11);
      return o;
    };
    P.b1 = blk(0); P.b2 = blk(12); P.b3 = blk(24);
    P.cls_b = F(36); P.cls_w = F(37); P.fc_b = F(38); P.fc_w = F(39);
    P.p1_b = F(40); P.p1_w = F(41); P.p2_b = F(42); P.p2_w = F(43);
    P.phi_im = F(44); P.phi_re = F(45); P.r_imag = F(46); P.r_real = F(47);
  }

  char* w8 = (char*)d_ws;
  float*  stepp = (float*)(w8);
  __bf16* bf1r = (__bf16*)(w8 + O_BF1R);
  __bf16* bf1i = (__bf16*)(w8 + O_BF1I);
  __bf16* bf2r = (__bf16*)(w8 + O_BF2R);
  __bf16* bf2i = (__bf16*)(w8 + O_BF2I);
  float* h    = (float*)(w8 + O_H);
  float* buf1 = (float*)(w8 + O_BUF1);
  float* buf2 = (float*)(w8 + O_BUF2);
  float* buf3 = (float*)(w8 + O_BUF3);
  float* stats = (float*)(w8 + O_STAT);
  float* pooled = (float*)(w8 + O_POOL);
  float* q  = (float*)(w8 + O_Q);
  float* z1 = (float*)(w8 + O_Z1);
  float* z2 = (float*)(w8 + O_Z2);
  float* outp = (float*)d_out;

  k_spectral<<<1, 512, 0, stream>>>(P.phi_re, P.phi_im, stepp);
  {
    int total = KT1 * NT1 * 512 + KT2 * NT2 * 512;
    k_prep_frags<<<(total + 255) / 256, 256, 0, stream>>>(P.phi_re, P.phi_im,
                                                          bf1r, bf1i, bf2r, bf2i);
  }
  {
    size_t ldsBytes = (size_t)(16 * NPAD * 2 + 16 * MPAD * 2) * sizeof(float)
                    + (size_t)(16 * MPAD * 2) * sizeof(__bf16);   // 192512 B
    k_ista<<<BATCH / 16, 512, ldsBytes, stream>>>(P.r_real, P.r_imag,
                                                  bf1r, bf1i, bf2r, bf2i, stepp, h);
  }

  auto launch_conv = [&](const float* x, const float* wgt, const float* b, float* y,
                         int Cin, int Lin, int Cout, int Lout, int k, int s) {
    int K = Cin * k;
    int ktiles = (K + 31) / 32;
    int Kpad = ktiles * 32;
    int tilesN = (Cout + 15) / 16;
    int Mtot = BATCH * Lout;
    int tilesM = (Mtot + 15) / 16;
    int mtGroups = (tilesM + 31) / 32;
    int blocks = mtGroups * tilesN;
    size_t smem = (size_t)16 * Kpad * 4;
    if (k == 5)      k_conv<5><<<blocks, 256, smem, stream>>>(x, wgt, b, y, Cin, Lin, Cout, Lout, s);
    else if (k == 7) k_conv<7><<<blocks, 256, smem, stream>>>(x, wgt, b, y, Cin, Lin, Cout, Lout, s);
    else             k_conv<1><<<blocks, 256, smem, stream>>>(x, wgt, b, y, Cin, Lin, Cout, Lout, s);
  };
  auto run_stats = [&](const float* y, int C, int L, int set) {
    float* s1 = stats + set * 512;
    k_zero<<<2, 256, 0, stream>>>(s1, 512);
    k_stats<<<256, 256, 0, stream>>>(y, s1, s1 + 256, C, L);
  };
  auto bn1 = [&](const float* a, int set, const float* g, const float* be,
                 float* out, int C, int L, int relu) {
    float* s1 = stats + set * 512;
    k_bn_apply<<<1024, 256, 0, stream>>>(a, s1, s1 + 256, g, be,
                                         nullptr, nullptr, nullptr, nullptr, nullptr,
                                         out, C, L, relu);
  };
  auto bn2 = [&](const float* a, int sa, const float* ga, const float* ba_,
                 const float* b2p, int sb, const float* gb, const float* bb_,
                 float* out, int C, int L) {
    float* sA = stats + sa * 512; float* sB = stats + sb * 512;
    k_bn_apply<<<1024, 256, 0, stream>>>(a, sA, sA + 256, ga, ba_,
                                         b2p, sB, sB + 256, gb, bb_, out, C, L, 1);
  };
  auto block = [&](const float* xin, int Cin, int Lin, int stride, int k, const BlkP& BP,
                   int Cout, int Lout, float* bY1, float* bY2, float* outbuf) {
    launch_conv(xin, BP.w1, BP.b1, bY1, Cin, Lin, Cout, Lout, k, stride);
    run_stats(bY1, Cout, Lout, 0);
    bn1(bY1, 0, BP.g1, BP.be1, bY1, Cout, Lout, 1);
    launch_conv(bY1, BP.w2, BP.b2, bY2, Cout, Lout, Cout, Lout, k, 1);
    launch_conv(xin, BP.ws, BP.bs, bY1, Cin, Lin, Cout, Lout, 1, stride);
    run_stats(bY2, Cout, Lout, 0);
    run_stats(bY1, Cout, Lout, 1);
    bn2(bY2, 0, BP.g2, BP.be2, bY1, 1, BP.gs, BP.bes, outbuf, Cout, Lout);
  };

  block(h,    1,   984, 1, 5, P.b1,  64, 984, buf2, buf3, buf1);
  block(buf1, 64,  984, 2, 7, P.b2, 128, 492, buf2, buf3, buf1);
  block(buf1, 128, 492, 2, 5, P.b3, 256, 246, buf2, buf3, buf1);

  k_pool<<<(BATCH * 256 + 255) / 256, 256, 0, stream>>>(buf1, pooled, 256, 246);

  auto launch_gemm = [&](const float* X, const float* W, const float* b, float* Cp,
                         int M, int N, int K, int relu) {
    long tiles = (long)((M + 15) / 16) * ((N + 15) / 16);
    int blocks = (int)((tiles + 7) / 8);
    k_gemm<<<blocks, 256, 0, stream>>>(X, W, b, Cp, M, N, K, relu);
  };

  launch_gemm(pooled, P.fc_w, P.fc_b, q, BATCH, 1024, 256, 0);
  k_l2norm<<<BATCH, 256, 0, stream>>>(q, q, 1024);
  launch_gemm(q, P.cls_w, P.cls_b, outp, BATCH, 10, 1024, 0);
  launch_gemm(q, P.p1_w, P.p1_b, z1, BATCH, 512, 1024, 1);
  launch_gemm(z1, P.p2_w, P.p2_b, z2, BATCH, 128, 512, 0);
  k_l2norm<<<BATCH, 256, 0, stream>>>(z2, outp + BATCH * 10, 128);

  (void)out_size; (void)ws_size; (void)n_in;
}
